// VectorQuantizer_36438502540044
// MI455X (gfx1250) — compile-verified, matrix operand order checked by reading, not run on hardware
//
#include <hip/hip_runtime.h>
#include <hip/hip_bf16.h>

typedef __attribute__((ext_vector_type(16))) __bf16 v16bf;
typedef __attribute__((ext_vector_type(8)))  float  v8f;

#define D_DIM 64
#define K_DIM 512
#define ROWS_PER_BLOCK 128
#define LDS_STRIDE 72   // padded bf16 row stride: 36 dwords -> conflict-free row access

union BOp { v16bf v; uint4 q[2]; };

#define CVT8(a, base, f0, f1)                                                  \
  a[base+0]=(__bf16)f0.x; a[base+1]=(__bf16)f0.y;                              \
  a[base+2]=(__bf16)f0.z; a[base+3]=(__bf16)f0.w;                              \
  a[base+4]=(__bf16)f1.x; a[base+5]=(__bf16)f1.y;                              \
  a[base+6]=(__bf16)f1.z; a[base+7]=(__bf16)f1.w;

__global__ __launch_bounds__(256) void vq_main(const float* __restrict__ enc,
                                               const float* __restrict__ wgt,
                                               float* __restrict__ idxOut,
                                               float* __restrict__ qOut,
                                               float* __restrict__ partial)
{
  __shared__ __align__(16) __bf16 sW[K_DIM * LDS_STRIDE];   // 73728 B
  __shared__ float sW2[K_DIM];                              // 2048 B
  __shared__ int   sIdx[ROWS_PER_BLOCK];                    // 512 B
  __shared__ float sRed[256];                               // 1024 B

  const int tid  = threadIdx.x;
  const int lane = tid & 31;
  const int wave = tid >> 5;
  const int blockRow0 = blockIdx.x * ROWS_PER_BLOCK;

  // ---- stage codebook into LDS as bf16 + per-row squared norms (f32) ----
  {
    unsigned* sWu = (unsigned*)sW;
    #pragma unroll
    for (int rr = 0; rr < 2; ++rr) {
      const int r = tid * 2 + rr;
      const float4* wp = (const float4*)(wgt + (size_t)r * D_DIM);
      float ss = 0.f;
      const unsigned rowBaseU = (unsigned)(r * LDS_STRIDE) >> 1;
      #pragma unroll
      for (int v = 0; v < D_DIM / 4; ++v) {
        float4 f = wp[v];
        ss = __builtin_fmaf(f.x, f.x, ss); ss = __builtin_fmaf(f.y, f.y, ss);
        ss = __builtin_fmaf(f.z, f.z, ss); ss = __builtin_fmaf(f.w, f.w, ss);
        unsigned p0 = (unsigned)__builtin_bit_cast(unsigned short, (__bf16)f.x) |
                      ((unsigned)__builtin_bit_cast(unsigned short, (__bf16)f.y) << 16);
        unsigned p1 = (unsigned)__builtin_bit_cast(unsigned short, (__bf16)f.z) |
                      ((unsigned)__builtin_bit_cast(unsigned short, (__bf16)f.w) << 16);
        sWu[rowBaseU + v * 2 + 0] = p0;
        sWu[rowBaseU + v * 2 + 1] = p1;
      }
      sW2[r] = ss;
    }
  }
  __syncthreads();

  // ---- load this wave's 16-row x-tile into WMMA A layout (bf16), kept in regs ----
  const int r0   = blockRow0 + wave * 16;
  const int mrow = r0 + (lane & 15);
  const int koff = (lane < 16) ? 0 : 8;   // K sub-block per lane half (ISA A layout)

  v16bf a0, a1;
  {
    const float* rowp = enc + (size_t)mrow * D_DIM + koff;
    {
      const float4* p = (const float4*)(rowp);        // k = koff..+7, koff+16..+23
      float4 c0 = p[0], c1 = p[1], c2 = p[4], c3 = p[5];
      CVT8(a0, 0, c0, c1);
      CVT8(a0, 8, c2, c3);
    }
    {
      const float4* p = (const float4*)(rowp + 32);   // k = 32+koff..., 48+koff...
      float4 c0 = p[0], c1 = p[1], c2 = p[4], c3 = p[5];
      CVT8(a1, 0, c0, c1);
      CVT8(a1, 8, c2, c3);
    }
  }

  // ---- sweep K=512 codes, 2 column tiles (32 cols) per iteration ----
  float bestV[8];
  int   bestI[8];
  #pragma unroll
  for (int i = 0; i < 8; ++i) { bestV[i] = 3.4e38f; bestI[i] = 0; }

  const int bcol = lane & 15;
  for (int ct = 0; ct < K_DIM / 16; ct += 2) {
    const int colA = ct * 16 + bcol;
    const int colB = colA + 16;

    const __bf16* wrowA = sW + colA * LDS_STRIDE + koff;
    const __bf16* wrowB = sW + colB * LDS_STRIDE + koff;
    BOp bA0, bA1, bB0, bB1;
    bA0.q[0] = *(const uint4*)(wrowA);
    bA0.q[1] = *(const uint4*)(wrowA + 16);
    bA1.q[0] = *(const uint4*)(wrowA + 32);
    bA1.q[1] = *(const uint4*)(wrowA + 48);
    bB0.q[0] = *(const uint4*)(wrowB);
    bB0.q[1] = *(const uint4*)(wrowB + 16);
    bB1.q[0] = *(const uint4*)(wrowB + 32);
    bB1.q[1] = *(const uint4*)(wrowB + 48);

    v8f cA = {};
    v8f cB = {};
    cA = __builtin_amdgcn_wmma_f32_16x16x32_bf16(false, a0, false, bA0.v,
                                                 (short)0, cA, false, false);
    cB = __builtin_amdgcn_wmma_f32_16x16x32_bf16(false, a0, false, bB0.v,
                                                 (short)0, cB, false, false);
    cA = __builtin_amdgcn_wmma_f32_16x16x32_bf16(false, a1, false, bA1.v,
                                                 (short)0, cA, false, false);
    cB = __builtin_amdgcn_wmma_f32_16x16x32_bf16(false, a1, false, bB1.v,
                                                 (short)0, cB, false, false);

    const float w2A = sW2[colA];
    const float w2B = sW2[colB];
    #pragma unroll
    for (int i = 0; i < 8; ++i) {
      float sA = __builtin_fmaf(-2.f, cA[i], w2A);  // ||w||^2 - 2 x.w
      if (sA < bestV[i]) { bestV[i] = sA; bestI[i] = colA; }
    }
    #pragma unroll
    for (int i = 0; i < 8; ++i) {
      float sB = __builtin_fmaf(-2.f, cB[i], w2B);
      if (sB < bestV[i]) { bestV[i] = sB; bestI[i] = colB; }
    }
  }

  // ---- argmin reduce across the 16 columns held in each lane half ----
  #pragma unroll
  for (int i = 0; i < 8; ++i) {
    #pragma unroll
    for (int off = 8; off >= 1; off >>= 1) {
      float ov = __shfl_xor(bestV[i], off, 32);
      int   oi = __shfl_xor(bestI[i], off, 32);
      if (ov < bestV[i] || (ov == bestV[i] && oi < bestI[i])) {
        bestV[i] = ov; bestI[i] = oi;
      }
    }
  }
  if ((lane & 15) == 0) {
    const int half = (lane >> 4) ? 8 : 0;   // C layout: lanes>=16 hold rows M=8..15
    #pragma unroll
    for (int i = 0; i < 8; ++i) {
      sIdx[wave * 16 + half + i] = bestI[i];
      idxOut[r0 + half + i] = (float)bestI[i];
    }
  }
  __syncthreads();

  // ---- gather quantized = W[idx] in full f32, write out, accumulate (q-x)^2 ----
  float lsum = 0.f;
  for (int e = tid; e < ROWS_PER_BLOCK * D_DIM; e += 256) {
    const int row = e >> 6;
    const int kk  = e & 63;
    const int id  = sIdx[row];
    const float wv = wgt[(size_t)id * D_DIM + kk];
    const size_t g = (size_t)(blockRow0 + row) * D_DIM + kk;
    const float xv = enc[g];
    qOut[g] = wv;
    const float d = wv - xv;
    lsum = __builtin_fmaf(d, d, lsum);
  }
  sRed[tid] = lsum;
  __syncthreads();
  for (int s = 128; s > 0; s >>= 1) {
    if (tid < s) sRed[tid] += sRed[tid + s];
    __syncthreads();
  }
  if (tid == 0) partial[blockIdx.x] = sRed[0];
}

__global__ __launch_bounds__(256) void vq_loss_reduce(const float* __restrict__ partial,
                                                      int nb,
                                                      float* __restrict__ lossOut,
                                                      float scale)
{
  __shared__ float sRed[256];
  float s = 0.f;
  for (int i = threadIdx.x; i < nb; i += 256) s += partial[i];
  sRed[threadIdx.x] = s;
  __syncthreads();
  for (int t = 128; t > 0; t >>= 1) {
    if (threadIdx.x < t) sRed[threadIdx.x] += sRed[threadIdx.x + t];
    __syncthreads();
  }
  if (threadIdx.x == 0) *lossOut = sRed[0] * scale;
}

extern "C" void kernel_launch(void* const* d_in, const int* in_sizes, int n_in,
                              void* d_out, int out_size, void* d_ws, size_t ws_size,
                              hipStream_t stream) {
  const float* enc = (const float*)d_in[0];   // [N, 64] f32
  const float* wgt = (const float*)d_in[1];   // [512, 64] f32
  const int N = in_sizes[0] / D_DIM;          // 262144

  float* out     = (float*)d_out;
  float* idxOut  = out;                         // N (indices stored as float values)
  float* qOut    = out + N;                     // N*64
  float* lossOut = out + N + (size_t)N * D_DIM; // 1

  float* partial = (float*)d_ws;              // nb floats of scratch
  const int nb = N / ROWS_PER_BLOCK;          // 2048 blocks

  vq_main<<<nb, 256, 0, stream>>>(enc, wgt, idxOut, qOut, partial);

  const float scale = 1.25f / ((float)N * (float)D_DIM); // (1 + BETA) / (N*D)
  vq_loss_reduce<<<1, 256, 0, stream>>>(partial, nb, lossOut, scale);
}
